// L1_17738214932834
// MI455X (gfx1250) — matrix operand order repaired and checked by reading, not compiled, not run on hardware
//
#include <hip/hip_runtime.h>
#include <hip/hip_bf16.h>
#include <math.h>

// ---------------- model dims ----------------
#define Bn 8
#define Tn 1024
#define Dm 512
#define Hn 8
#define DH 64
#define DEPTH 8
#define FF 2048
#define NH 8
#define BS 64
#define NB 16
#define NC (NH*NB)      // 128 chunks
#define BH (Bn*Hn)      // 64
#define NHT (NH*Tn)     // 8192

typedef __attribute__((ext_vector_type(16))) __bf16 v16bf;
typedef __attribute__((ext_vector_type(8)))  float  v8f;

union BF16x16 { v16bf v; unsigned short u[16]; uint4 q[2]; };

__device__ __forceinline__ unsigned short f2bf(float f) {
    unsigned int u = __float_as_uint(f);
    unsigned int r = u + 0x7FFFu + ((u >> 16) & 1u);
    return (unsigned short)(r >> 16);
}

__device__ __forceinline__ float gelu_tanh(float x) {
    float x3 = x * x * x;
    return 0.5f * x * (1.0f + tanhf(0.7978845608028654f * (x + 0.044715f * x3)));
}

// ---------- CDNA5 async global->LDS staging (guarded; uint4 fallback) -----
#if defined(__has_builtin)
#if __has_builtin(__builtin_amdgcn_global_load_async_to_lds_b128)
#define HAVE_ASYNC_LDS 1
#endif
#endif

#ifdef HAVE_ASYNC_LDS
typedef int nv4i __attribute__((vector_size(16)));   // native clang vector, not HIP_vector_type
typedef __attribute__((address_space(1))) nv4i as1_v4i;
typedef __attribute__((address_space(3))) nv4i as3_v4i;
__device__ __forceinline__ void async_copy16(const unsigned short* g, unsigned short* l) {
    __builtin_amdgcn_global_load_async_to_lds_b128(
        (as1_v4i*)(void*)g, (as3_v4i*)(void*)l, 0, 0);
}
__device__ __forceinline__ void async_wait0() {
#if __has_builtin(__builtin_amdgcn_s_wait_asynccnt)
    __builtin_amdgcn_s_wait_asynccnt(0);
#else
    asm volatile("s_wait_asynccnt 0" ::: "memory");
#endif
}
#endif

// ======================================================================
// GEMM: C[M,N](f32) (=|+=) act(A[M,K](bf16) @ Bsw[K,N](bf16,swizzled) + bias)
// Bsw layout: [K/32][N/16][n:16][k':32]  (per 32x16 block, column-major)
//   -> per K-step B tile is one contiguous 8KB region
//   -> B fragment = two 16-byte runs 32 bytes apart = 2x ds_load_b128
// block 256 = 8 waves; tile 128x128x32; wave -> 64x32 (4x2 wmma tiles)
// ======================================================================
template<bool BIAS, bool GELU_, bool RES, bool WF32, bool WBF16>
__global__ __launch_bounds__(256) void gemm_bf16_kernel(
    const unsigned short* __restrict__ A, const unsigned short* __restrict__ Bsw,
    const float* __restrict__ bias, float* __restrict__ Cf,
    unsigned short* __restrict__ Cbf, int M, int N, int K)
{
    __shared__ __align__(16) unsigned short sA[128 * 32];
    __shared__ __align__(16) unsigned short sB[32 * 128];

    const int tid  = threadIdx.x;
    const int lane = tid & 31;
    const int w    = tid >> 5;
    const int wm   = w >> 2;   // 0..1  -> 64-row half
    const int wn   = w & 3;    // 0..3  -> 32-col quarter
    const int loff = lane >> 4;
    const int l15  = lane & 15;
    const int bM   = blockIdx.y * 128;
    const int bN   = blockIdx.x * 128;
    const int ng   = N >> 4;

    v8f acc[4][2];
    for (int i = 0; i < 4; i++)
        for (int j = 0; j < 2; j++)
            for (int e = 0; e < 8; e++) acc[i][j][e] = 0.0f;

    for (int k0 = 0; k0 < K; k0 += 32) {
        const size_t btile = ((size_t)(k0 >> 5) * ng + (bN >> 4)) * 512;
#ifdef HAVE_ASYNC_LDS
        for (int c = tid; c < 512; c += 256) {
            int row = c >> 2, cc = (c & 3) * 8;
            async_copy16(A + (size_t)(bM + row) * K + k0 + cc, &sA[row * 32 + cc]);
            async_copy16(Bsw + btile + (size_t)c * 8, &sB[c * 8]);
        }
        async_wait0();
#else
        for (int c = tid; c < 512; c += 256) {
            int row = c >> 2, cc = (c & 3) * 8;
            *(uint4*)&sA[row * 32 + cc] =
                *(const uint4*)(A + (size_t)(bM + row) * K + k0 + cc);
            *(uint4*)&sB[c * 8] = *(const uint4*)(Bsw + btile + (size_t)c * 8);
        }
#endif
        if (k0 + 32 < K) {  // CDNA5 global_prefetch_b8 of next tiles
            int row = tid >> 2, cc = (tid & 3) * 8;
            __builtin_prefetch(A + (size_t)(bM + row) * K + k0 + 32 + cc, 0, 1);
            __builtin_prefetch(Bsw + btile + (size_t)ng * 512 + tid * 8, 0, 1);
        }
        __syncthreads();

        BF16x16 af[4], bfr[2];
        for (int rm = 0; rm < 4; rm++) {
            int m = wm * 64 + rm * 16 + l15;
            af[rm].q[0] = *(const uint4*)&sA[m * 32 + loff * 8];
            af[rm].q[1] = *(const uint4*)&sA[m * 32 + 16 + loff * 8];
        }
        for (int cn = 0; cn < 2; cn++) {
            int o = ((wn * 2 + cn) * 16 + l15) * 32 + loff * 8;
            bfr[cn].q[0] = *(const uint4*)&sB[o];
            bfr[cn].q[1] = *(const uint4*)&sB[o + 16];
        }
        for (int rm = 0; rm < 4; rm++)
            for (int cn = 0; cn < 2; cn++)
                acc[rm][cn] = __builtin_amdgcn_wmma_f32_16x16x32_bf16(
                    false, af[rm].v, false, bfr[cn].v, (short)0, acc[rm][cn], false, false);
        __syncthreads();
    }

    for (int rm = 0; rm < 4; rm++)
        for (int cn = 0; cn < 2; cn++) {
            int col = bN + wn * 32 + cn * 16 + l15;
            float bv = BIAS ? bias[col] : 0.0f;
            for (int r = 0; r < 8; r++) {
                int row = bM + wm * 64 + rm * 16 + r + (loff ? 8 : 0);
                float val = acc[rm][cn][r];
                if (BIAS)  val += bv;
                if (GELU_) val = gelu_tanh(val);
                size_t idx = (size_t)row * N + col;
                if (RES)        Cf[idx] += val;
                else if (WF32)  Cf[idx]  = val;
                if (WBF16)      Cbf[idx] = f2bf(val);
            }
        }
}

// ======================================================================
// embedding
// ======================================================================
__global__ void embed_kernel(const int* __restrict__ ids, const float* __restrict__ tok,
                             const float* __restrict__ pos, float* __restrict__ x1,
                             float* __restrict__ x2)
{
    size_t i = (size_t)blockIdx.x * 256 + threadIdx.x;  // B*T*D
    int d = (int)(i & (Dm - 1));
    size_t bt = i >> 9;
    int t = (int)(bt & (Tn - 1));
    float v = tok[(size_t)ids[bt] * Dm + d] + pos[(size_t)t * Dm + d];
    x1[i] = v; x2[i] = v;
}

// ======================================================================
// layernorm: one wave32 per 512-wide row; writes bf16 (+ optional f32)
// ======================================================================
template<bool WF32>
__global__ __launch_bounds__(256) void ln_kernel(
    const float* __restrict__ x, const float* __restrict__ g, const float* __restrict__ b,
    unsigned short* __restrict__ obf, float* __restrict__ of)
{
    int row  = blockIdx.x * 8 + (threadIdx.x >> 5);
    int lane = threadIdx.x & 31;
    const float* xr = x + (size_t)row * Dm;
    float v[16], s = 0.0f;
    for (int i = 0; i < 16; i++) { v[i] = xr[lane + i * 32]; s += v[i]; }
    for (int m = 16; m >= 1; m >>= 1) s += __shfl_xor(s, m, 32);
    float mean = s * (1.0f / Dm);
    float q = 0.0f;
    for (int i = 0; i < 16; i++) { float d = v[i] - mean; q += d * d; }
    for (int m = 16; m >= 1; m >>= 1) q += __shfl_xor(q, m, 32);
    float rstd = rsqrtf(q * (1.0f / Dm) + 1e-5f);
    for (int i = 0; i < 16; i++) {
        int col = lane + i * 32;
        float o = (v[i] - mean) * rstd * g[col] + b[col];
        obf[(size_t)row * Dm + col] = f2bf(o);
        if (WF32) of[(size_t)row * Dm + col] = o;
    }
}

// ======================================================================
// LSH hashing: bucket = argmax over [rotated, -rotated], per hash round
// ======================================================================
__global__ void hash_kernel(const float* __restrict__ qk, const float* __restrict__ rot,
                            int* __restrict__ buckets)
{
    int id = blockIdx.x * 256 + threadIdx.x;  // BH*T = 65536
    int bh = id >> 10, t = id & (Tn - 1);
    const float* qr = qk + ((size_t)bh * Tn + t) * DH;
    float acc[NH][8];
    for (int nh = 0; nh < NH; nh++) for (int r = 0; r < 8; r++) acc[nh][r] = 0.0f;
#pragma unroll 4
    for (int d = 0; d < DH; d++) {
        float f = qr[d];
        const float* rr = rot + d * (NH * 8);
        for (int nh = 0; nh < NH; nh++)
            for (int r = 0; r < 8; r++) acc[nh][r] += f * rr[nh * 8 + r];
    }
    for (int nh = 0; nh < NH; nh++) {
        float best = acc[nh][0]; int bi = 0;
        for (int i = 1; i < 16; i++) {
            float vv = (i < 8) ? acc[nh][i] : -acc[nh][i - 8];
            if (vv > best) { best = vv; bi = i; }
        }
        buckets[(size_t)bh * NHT + nh * Tn + t] = nh * NB + bi;
    }
}

// ======================================================================
// stable counting sort per (bh): 8192 items, 128 bins. 64 threads.
// ======================================================================
__global__ __launch_bounds__(64) void sort_kernel(const int* __restrict__ buckets,
                                                  int* __restrict__ sticker)
{
    __shared__ int cnt[128 * 64];
    int bh = blockIdx.x, th = threadIdx.x;
    for (int i = th; i < 128 * 64; i += 64) cnt[i] = 0;
    __syncthreads();
    const int* bk = buckets + (size_t)bh * NHT;
    int base = th * 128;
    for (int i = 0; i < 128; i++) { int b = bk[base + i]; cnt[b * 64 + th]++; }
    __syncthreads();
    if (th == 0) {
        int run = 0;
        for (int i = 0; i < 128 * 64; i++) { int c = cnt[i]; cnt[i] = run; run += c; }
    }
    __syncthreads();
    int* out = sticker + (size_t)bh * NHT;
    for (int i = 0; i < 128; i++) {
        int j = base + i; int b = bk[j];
        int pos = cnt[b * 64 + th]++;
        out[pos] = j;
    }
}

// ======================================================================
// chunked LSH attention, one block per (chunk, bh). 8 waves, WMMA bf16.
// LDS operand layouts chosen so every WMMA fragment = 2x ds_load_b128:
//   sm_q    [q=64][dh=64]    (A frags, dots)
//   sm_k    [key=128][dh=64] (B frags, dots: N=key, K=dh contiguous)
//   sm_vT   [dh=64][key=128] (B frags, bo:   N=dh,  K=key contiguous)
//   sm_probs overwrites sm_k (A frags, bo)
// ======================================================================
__global__ __launch_bounds__(256) void lsh_attn_kernel(
    const float* __restrict__ qk, const float* __restrict__ vv,
    const int* __restrict__ sticker, float* __restrict__ o_rounds,
    float* __restrict__ lg_arr)
{
    extern __shared__ __align__(16) char smem[];
    float*          sm_dots = (float*)smem;                     // 64*128 f32 (32768B)
    unsigned short* sm_k    = (unsigned short*)(smem + 32768);  // [key][dh] / probs (16384B)
    unsigned short* sm_vT   = (unsigned short*)(smem + 49152);  // [dh][key]   (16384B)
    unsigned short* sm_q    = (unsigned short*)(smem + 65536);  // [q][dh]     (8192B)
    int*   sm_qj    = (int*)(smem + 73728);                     // 64
    int*   sm_qt    = sm_qj + 64;                               // 64
    int*   sm_kvt   = sm_qt + 64;                               // 128
    float* sm_scale = (float*)(sm_kvt + 128);                   // 128
    float* sm_lg    = sm_scale + 128;                           // 64

    const int c    = blockIdx.x;
    const int bh   = blockIdx.y;
    const int tid  = threadIdx.x;
    const int lane = tid & 31;
    const int w    = tid >> 5;
    const int loff = lane >> 4;
    const int l15  = lane & 15;

    if (tid < 64) {
        int j = sticker[(size_t)bh * NHT + c * BS + tid];
        sm_qj[tid] = j;
        int t = j & (Tn - 1);
        sm_qt[tid] = t; sm_kvt[tid] = t;
    } else if (tid < 128) {
        int i = tid - 64;
        int cp = (c + NC - 1) & (NC - 1);   // look-back one chunk (with wrap)
        int j = sticker[(size_t)bh * NHT + cp * BS + i];
        sm_kvt[64 + i] = j & (Tn - 1);
    }
    __syncthreads();

    if (tid < 128) {  // key row scales: 1/(||row|| + 1e-8)
        const float* kr = qk + ((size_t)bh * Tn + sm_kvt[tid]) * DH;
        float s = 0.0f;
        for (int d = 0; d < DH; d++) { float f = kr[d]; s += f * f; }
        sm_scale[tid] = 1.0f / (sqrtf(s) + 1e-8f);
    }
    __syncthreads();

    for (int e = tid; e < 64 * 64; e += 256) {        // q tile
        int i = e >> 6, d = e & 63;
        sm_q[i * 64 + d] = f2bf(qk[((size_t)bh * Tn + sm_qt[i]) * DH + d]);
    }
    for (int e = tid; e < 128 * 64; e += 256) {       // k and vT tiles
        int key = e >> 6, d = e & 63;
        int t = sm_kvt[key];
        sm_k[key * 64 + d]   = f2bf(qk[((size_t)bh * Tn + t) * DH + d] * sm_scale[key]);
        sm_vT[d * 128 + key] = f2bf(vv[((size_t)bh * Tn + t) * DH + d]);
    }
    __syncthreads();

    // ---- dots = q @ k^T  (64x128, K=64): wave -> 1 row-tile x 4 col-tiles
    {
        const int r  = w & 3;
        const int cg = (w >> 2) * 4;
        v8f acc[4];
        for (int cn = 0; cn < 4; cn++) for (int e = 0; e < 8; e++) acc[cn][e] = 0.0f;
        for (int kb = 0; kb < 64; kb += 32) {
            BF16x16 af;
            int m = r * 16 + l15;
            af.q[0] = *(const uint4*)&sm_q[m * 64 + kb + loff * 8];
            af.q[1] = *(const uint4*)&sm_q[m * 64 + kb + 16 + loff * 8];
            for (int cn = 0; cn < 4; cn++) {
                BF16x16 bfr;
                int o = ((cg + cn) * 16 + l15) * 64 + kb + loff * 8;
                bfr.q[0] = *(const uint4*)&sm_k[o];
                bfr.q[1] = *(const uint4*)&sm_k[o + 16];
                acc[cn] = __builtin_amdgcn_wmma_f32_16x16x32_bf16(
                    false, af.v, false, bfr.v, (short)0, acc[cn], false, false);
            }
        }
        for (int cn = 0; cn < 4; cn++)
            for (int rr = 0; rr < 8; rr++) {
                int m = r * 16 + rr + (loff ? 8 : 0);
                int n = (cg + cn) * 16 + l15;
                float val = acc[cn][rr] * 0.125f;          // DH^-0.5
                if (sm_qt[m] == sm_kvt[n]) val = -5e4f;    // self-mask
                sm_dots[m * 128 + n] = val;
            }
    }
    __syncthreads();

    // ---- rowwise logsumexp -> lg scatter; probs (bf16) overwrite sm_k
    if (tid < 64) {
        float mx = -1e30f;
        for (int k = 0; k < 128; k++) mx = fmaxf(mx, sm_dots[tid * 128 + k]);
        float s = 0.0f;
        for (int k = 0; k < 128; k++) s += expf(sm_dots[tid * 128 + k] - mx);
        float lgv = mx + logf(s);
        sm_lg[tid] = lgv;
        lg_arr[(size_t)bh * NHT + sm_qj[tid]] = lgv;
    }
    __syncthreads();
    for (int e = tid; e < 64 * 128; e += 256) {
        int m = e >> 7, k = e & 127;
        sm_k[m * 128 + k] = f2bf(expf(sm_dots[m * 128 + k] - sm_lg[m]));
    }
    __syncthreads();

    // ---- bo = probs @ v  (64x64, K=128): wave -> 1 row-tile x 2 col-tiles
    {
        const int r  = w & 3;
        const int c0 = (w >> 2) * 2;
        v8f acc[2];
        for (int cn = 0; cn < 2; cn++) for (int e = 0; e < 8; e++) acc[cn][e] = 0.0f;
        for (int kb = 0; kb < 128; kb += 32) {
            BF16x16 af;
            int m = r * 16 + l15;
            af.q[0] = *(const uint4*)&sm_k[m * 128 + kb + loff * 8];
            af.q[1] = *(const uint4*)&sm_k[m * 128 + kb + 16 + loff * 8];
            for (int cn = 0; cn < 2; cn++) {
                BF16x16 bfr;
                int o = ((c0 + cn) * 16 + l15) * 128 + kb + loff * 8;
                bfr.q[0] = *(const uint4*)&sm_vT[o];
                bfr.q[1] = *(const uint4*)&sm_vT[o + 16];
                acc[cn] = __builtin_amdgcn_wmma_f32_16x16x32_bf16(
                    false, af.v, false, bfr.v, (short)0, acc[cn], false, false);
            }
        }
        for (int cn = 0; cn < 2; cn++)
            for (int rr = 0; rr < 8; rr++) {
                int m = r * 16 + rr + (loff ? 8 : 0);
                int n = (c0 + cn) * 16 + l15;
                o_rounds[((size_t)bh * NHT + sm_qj[m]) * DH + n] = acc[cn][rr];
            }
    }
}

// ======================================================================
// combine hash rounds: softmax over per-round logsumexp, weighted sum.
// ======================================================================
__global__ __launch_bounds__(64) void combine_kernel(
    const float* __restrict__ o_rounds, const float* __restrict__ lg_arr,
    unsigned short* __restrict__ attn_bf)
{
    int bt = blockIdx.x;
    int bh = bt >> 10, t = bt & (Tn - 1);
    int d = threadIdx.x;
    float l[NH], mx = -1e30f;
    for (int nh = 0; nh < NH; nh++) {
        l[nh] = lg_arr[(size_t)bh * NHT + nh * Tn + t];
        mx = fmaxf(mx, l[nh]);
    }
    float s = 0.0f;
    for (int nh = 0; nh < NH; nh++) { l[nh] = expf(l[nh] - mx); s += l[nh]; }
    float inv = 1.0f / s, o = 0.0f;
    for (int nh = 0; nh < NH; nh++)
        o += l[nh] * inv * o_rounds[((size_t)bh * NHT + nh * Tn + t) * DH + d];
    int b = bh >> 3, h = bh & 7;
    attn_bf[((size_t)b * Tn + t) * Dm + h * DH + d] = f2bf(o);
}

// ======================================================================
// weight conversion f32 -> bf16 in WMMA-swizzled layout
// dst[kb][g][n][k']  (kb=K/32 blocks, g=N/16 groups, n=0..15, k'=0..31)
// ======================================================================
__global__ void convert_w_swizzle_kernel(const float* __restrict__ in,
                                         unsigned short* __restrict__ out,
                                         int N, int n)
{
    int i = blockIdx.x * 256 + threadIdx.x;
    if (i >= n) return;
    int kp = i & 31;
    int nn = (i >> 5) & 15;
    int rest = i >> 9;
    int ng = N >> 4;
    int g  = rest % ng;
    int kb = rest / ng;
    out[i] = f2bf(in[(size_t)(kb * 32 + kp) * N + g * 16 + nn]);
}

// ======================================================================
// small helpers
// ======================================================================
__global__ void halfsum_kernel(const float* __restrict__ a, const float* __restrict__ b,
                               float* __restrict__ o)
{
    size_t i = (size_t)blockIdx.x * 256 + threadIdx.x;
    o[i] = 0.5f * (a[i] + b[i]);
}

__global__ void mean_t_kernel(const float* __restrict__ x, float* __restrict__ o)
{
    int i = blockIdx.x * 256 + threadIdx.x;   // B*D = 4096
    int b = i >> 9, d = i & (Dm - 1);
    float s = 0.0f;
    for (int t = 0; t < Tn; t++) s += x[((size_t)b * Tn + t) * Dm + d];
    o[i] = s * (1.0f / Tn);
}

__global__ void fc_kernel(const float* __restrict__ m, const float* __restrict__ Wfc,
                          float* __restrict__ out)
{
    int i = blockIdx.x * 256 + threadIdx.x;   // B*D = 4096
    int b = i >> 9, j = i & (Dm - 1);
    float s = 0.0f;
    for (int d = 0; d < Dm; d++) s += m[b * Dm + d] * Wfc[(size_t)d * Dm + j];
    out[i] = s;
}

// ======================================================================
// host orchestration
// ======================================================================
extern "C" void kernel_launch(void* const* d_in, const int* in_sizes, int n_in,
                              void* d_out, int out_size, void* d_ws, size_t ws_size,
                              hipStream_t stream)
{
    const int*   ids      = (const int*)  d_in[0];
    const float* tok      = (const float*)d_in[1];
    const float* pos      = (const float*)d_in[2];
    const float* ln_a_g   = (const float*)d_in[3];
    const float* ln_a_b   = (const float*)d_in[4];
    const float* Wqk      = (const float*)d_in[5];
    const float* Wv       = (const float*)d_in[6];
    const float* Wo       = (const float*)d_in[7];
    const float* ln_f_g   = (const float*)d_in[8];   // ln_ff_g
    const float* ln_f_b   = (const float*)d_in[9];   // ln_ff_b
    const float* W1       = (const float*)d_in[10];
    const float* b1       = (const float*)d_in[11];
    const float* W2       = (const float*)d_in[12];
    const float* b2       = (const float*)d_in[13];
    const float* lnF_g    = (const float*)d_in[14];
    const float* lnF_b    = (const float*)d_in[15];
    const float* Wfc      = (const float*)d_in[16];
    const float* rotAll   = (const float*)d_in[17];
    float* out = (float*)d_out;

    // ---- workspace layout
    char* ws = (char*)d_ws;
    size_t off = 0;
    auto alloc = [&](size_t bytes) -> void* {
        void* p = ws + off;
        off = (off + bytes + 255) & ~(size_t)255;
        return p;
    };
    const size_t NTD = (size_t)Bn * Tn * Dm;            // 4M
    float* x1        = (float*)alloc(NTD * 4);
    float* x2        = (float*)alloc(NTD * 4);
    float* qk        = (float*)alloc(NTD * 4);
    float* vbuf      = (float*)alloc(NTD * 4);
    float* o_rounds  = (float*)alloc((size_t)BH * NHT * DH * 4);   // 128MB
    float* lg        = (float*)alloc((size_t)BH * NHT * 4);
    int*   buckets   = (int*)  alloc((size_t)BH * NHT * 4);
    int*   sticker   = (int*)  alloc((size_t)BH * NHT * 4);
    unsigned short* xn_bf   = (unsigned short*)alloc(NTD * 2);
    unsigned short* attn_bf = (unsigned short*)alloc(NTD * 2);
    unsigned short* h1_bf   = (unsigned short*)alloc((size_t)Bn * Tn * FF * 2);
    unsigned short* wbuf    = (unsigned short*)alloc((size_t)Dm * FF * 2);
    float* meanbuf   = (float*)alloc((size_t)Bn * Dm * 4);
    (void)ws_size; (void)n_in; (void)in_sizes; (void)out_size;

    const int M = Bn * Tn;                 // 8192 rows
    const size_t attn_smem = 75520;        // dots+k+vT+q+indices

    // embedding -> x1, x2
    embed_kernel<<<(unsigned)(NTD / 256), 256, 0, stream>>>(ids, tok, pos, x1, x2);

    for (int l = 0; l < DEPTH; l++) {
        const float* Wqk_l = Wqk + (size_t)l * Dm * Dm;
        const float* Wv_l  = Wv  + (size_t)l * Dm * Dm;
        const float* Wo_l  = Wo  + (size_t)l * Dm * Dm;
        const float* W1_l  = W1  + (size_t)l * Dm * FF;
        const float* W2_l  = W2  + (size_t)l * FF * Dm;
        const float* rot_l = rotAll + (size_t)l * DH * NH * (NB / 2);

        // xn = LN(x2) -> bf16
        ln_kernel<false><<<M / 8, 256, 0, stream>>>(x2, ln_a_g + l * Dm, ln_a_b + l * Dm,
                                                    xn_bf, nullptr);
        // qk = xn @ Wqk
        convert_w_swizzle_kernel<<<(Dm * Dm) / 256, 256, 0, stream>>>(Wqk_l, wbuf, Dm, Dm * Dm);
        gemm_bf16_kernel<false, false, false, true, false>
            <<<dim3(Dm / 128, M / 128), 256, 0, stream>>>(xn_bf, wbuf, nullptr, qk, nullptr,
                                                          M, Dm, Dm);
        // v = xn @ Wv
        convert_w_swizzle_kernel<<<(Dm * Dm) / 256, 256, 0, stream>>>(Wv_l, wbuf, Dm, Dm * Dm);
        gemm_bf16_kernel<false, false, false, true, false>
            <<<dim3(Dm / 128, M / 128), 256, 0, stream>>>(xn_bf, wbuf, nullptr, vbuf, nullptr,
                                                          M, Dm, Dm);
        // hash + stable sort
        hash_kernel<<<(BH * Tn) / 256, 256, 0, stream>>>(qk, rot_l, buckets);
        sort_kernel<<<BH, 64, 0, stream>>>(buckets, sticker);
        // chunked attention (WMMA) + scatter
        lsh_attn_kernel<<<dim3(NC, BH), 256, attn_smem, stream>>>(qk, vbuf, sticker,
                                                                  o_rounds, lg);
        // combine rounds -> attn_bf [B,T,D]
        combine_kernel<<<BH * Tn, 64, 0, stream>>>(o_rounds, lg, attn_bf);
        // x1 += attn @ Wo
        convert_w_swizzle_kernel<<<(Dm * Dm) / 256, 256, 0, stream>>>(Wo_l, wbuf, Dm, Dm * Dm);
        gemm_bf16_kernel<false, false, true, true, false>
            <<<dim3(Dm / 128, M / 128), 256, 0, stream>>>(attn_bf, wbuf, nullptr, x1, nullptr,
                                                          M, Dm, Dm);
        // h = LN(x1) -> bf16
        ln_kernel<false><<<M / 8, 256, 0, stream>>>(x1, ln_f_g + l * Dm, ln_f_b + l * Dm,
                                                    xn_bf, nullptr);
        // h1 = gelu(h @ W1 + b1) -> bf16
        convert_w_swizzle_kernel<<<(Dm * FF) / 256, 256, 0, stream>>>(W1_l, wbuf, FF, Dm * FF);
        gemm_bf16_kernel<true, true, false, false, true>
            <<<dim3(FF / 128, M / 128), 256, 0, stream>>>(xn_bf, wbuf, b1 + (size_t)l * FF,
                                                          nullptr, h1_bf, M, FF, Dm);
        // x2 += h1 @ W2 + b2
        convert_w_swizzle_kernel<<<(FF * Dm) / 256, 256, 0, stream>>>(W2_l, wbuf, Dm, FF * Dm);
        gemm_bf16_kernel<true, false, true, true, false>
            <<<dim3(Dm / 128, M / 128), 256, 0, stream>>>(h1_bf, wbuf, b2 + (size_t)l * Dm,
                                                          x2, nullptr, M, Dm, FF);
    }

    // final: x = 0.5*(x1+x2); LN; mean over T; @ Wfc
    halfsum_kernel<<<(unsigned)(NTD / 256), 256, 0, stream>>>(x1, x2, qk);
    ln_kernel<true><<<M / 8, 256, 0, stream>>>(qk, lnF_g, lnF_b, xn_bf, vbuf);
    mean_t_kernel<<<(Bn * Dm) / 256, 256, 0, stream>>>(vbuf, meanbuf);
    fc_kernel<<<(Bn * Dm) / 256, 256, 0, stream>>>(meanbuf, Wfc, out);
}
